// GATWithTransformerFusion_2723009266183
// MI455X (gfx1250) — compile-verified, matrix operand order checked by reading, not run on hardware
//
#include <hip/hip_runtime.h>
#include <hip/hip_bf16.h>
#include <math.h>

// ---------------------------------------------------------------------------
// Model constants (match reference)
// ---------------------------------------------------------------------------
#define NN       50000
#define EE       600000
#define IN_DIM   2000
#define HID      64
#define OUT_DIM  25
#define HEADS    4
#define NLAYERS  2
#define FFDIM    2048

#define CDIV(a, b) (((a) + (b) - 1) / (b))

typedef __attribute__((ext_vector_type(16))) __bf16 v16bf;
typedef __attribute__((ext_vector_type(8)))  float  v8f;

__device__ __forceinline__ int imin(int a, int b) { return a < b ? a : b; }

__device__ __forceinline__ unsigned short f2bf(float x) {
    union { float f; unsigned u; } c; c.f = x;
    unsigned r = c.u + 0x7FFFu + ((c.u >> 16) & 1u);   // round-to-nearest-even
    return (unsigned short)(r >> 16);
}

// bf16 fragment from an LDS tile stored [row][k] (kstride bf16 elems per row).
// CDNA5 16-bit A-layout: lane%16 = row; K pairs [pi0..pi0+3] and [pi0+8..pi0+11],
// pi0 = (lane>>4)*4.  B tiles stored [n][k] use the same loader.
__device__ __forceinline__ v16bf load_frag(const unsigned short* tile, int kstride,
                                           int row_off, int kbase, int lane) {
    int r   = row_off + (lane & 15);
    int pi0 = (lane >> 4) * 4;
    const unsigned* p = (const unsigned*)(tile + r * kstride + kbase);
    union { v16bf v; unsigned u[8]; } f;
    f.u[0] = p[pi0 + 0];  f.u[1] = p[pi0 + 1];
    f.u[2] = p[pi0 + 2];  f.u[3] = p[pi0 + 3];
    f.u[4] = p[pi0 + 8];  f.u[5] = p[pi0 + 9];
    f.u[6] = p[pi0 + 10]; f.u[7] = p[pi0 + 11];
    return f.v;
}

// bf16 fragment built from an f32 LDS tile stored [row][32] (A path: the tile
// was async-copied from global in memory order, converted here at frag build).
__device__ __forceinline__ v16bf load_frag_f32(const float* tile, int row_off, int lane) {
    int r  = row_off + (lane & 15);
    int kq = (lane >> 4) * 8;
    const float* p = tile + r * 32;
    union { v16bf v; unsigned short s[16]; } f;
#pragma unroll
    for (int t = 0; t < 4; ++t) {
        f.s[2 * t]         = f2bf(p[kq + 2 * t]);
        f.s[2 * t + 1]     = f2bf(p[kq + 2 * t + 1]);
        f.s[8 + 2 * t]     = f2bf(p[16 + kq + 2 * t]);
        f.s[8 + 2 * t + 1] = f2bf(p[16 + kq + 2 * t + 1]);
    }
    return f.v;
}

// ---------------------------------------------------------------------------
// GEMM: C[M,N] = act(A[M,K] @ B[K,N] + bias), f32 in/out, bf16 WMMA inside.
// Block: 256 threads (8 waves), tile BM x BN x 32.  grid.x == 1 in all uses
// (BN == N rounded up) so A is read exactly once from HBM.
// A tile staged in LDS as f32 in memory order via global_load_async_to_lds_b128
// (ASYNCcnt); B tile converted to bf16 [n][k].  Fringe rows/cols are clamped
// (results never stored); only the K-tail uses a guarded zero-fill path.
// ---------------------------------------------------------------------------
template <int BM, int BN>
__global__ __launch_bounds__(256)
void gemm_bf16_k(const float* __restrict__ A, const float* __restrict__ B,
                 float* __restrict__ C, const float* __restrict__ bias,
                 int M, int N, int K, int lda, int ldb, int ldc, int act) {
    constexpr int RW  = BM / 16;        // row wave-groups
    constexpr int CW  = 8 / RW;         // col wave-groups
    constexpr int ACC = BN / (16 * CW); // accumulators per wave

    __shared__ __attribute__((aligned(16))) float          As[BM * 32];
    __shared__ __attribute__((aligned(16))) unsigned short Bs[BN * 32];

    const int tid  = threadIdx.x;
    const int lane = tid & 31;
    const int w    = tid >> 5;
    const int wm   = w % RW;
    const int wn   = w / RW;
    const int m0   = blockIdx.y * BM;
    const int n0   = blockIdx.x * BN;

    v8f acc[ACC];
#pragma unroll
    for (int a = 0; a < ACC; ++a) acc[a] = (v8f){};

    for (int k0 = 0; k0 < K; k0 += 32) {
        const bool fullk = (k0 + 32 <= K);

        if (fullk) {
            // ---- A tile: async copy, f32, memory order; rows clamped ----
#pragma unroll
            for (int c = tid; c < BM * 8; c += 256) {   // 16B chunks
                int r  = c >> 3;
                int kq = (c & 7) * 4;
                int re = imin(m0 + r, M - 1);
                unsigned long long ga =
                    (unsigned long long)(A + (long)re * lda + k0 + kq);
                unsigned lo = (unsigned)(unsigned long long)(&As[r * 32 + kq]);
                asm volatile("global_load_async_to_lds_b128 %0, %1, off"
                             :: "v"(lo), "v"(ga) : "memory");
            }
            // ---- B tile: f32->bf16, [n][k]; coalesced along n; n clamped ----
#pragma unroll 2
            for (int i = tid; i < BN * 32; i += 256) {
                int n  = i & (BN - 1);
                int k  = i >> (BN == 256 ? 8 : (BN == 64 ? 6 : 5));
                int ne = imin(n0 + n, N - 1);
                Bs[n * 32 + k] = f2bf(B[(long)(k0 + k) * ldb + ne]);
            }
            asm volatile("s_wait_asynccnt 0x0" ::: "memory");
        } else {
            // K tail (only GEMM-1: 2000 % 32 == 16): zero-fill out-of-range K
            for (int i = tid; i < BM * 32; i += 256) {
                int r = i >> 5, k = i & 31;
                int re = imin(m0 + r, M - 1);
                As[i] = (k0 + k < K) ? A[(long)re * lda + k0 + k] : 0.f;
            }
            for (int i = tid; i < BN * 32; i += 256) {
                int n  = i & (BN - 1);
                int k  = i >> (BN == 256 ? 8 : (BN == 64 ? 6 : 5));
                int ne = imin(n0 + n, N - 1);
                float v = (k0 + k < K) ? B[(long)(k0 + k) * ldb + ne] : 0.f;
                Bs[n * 32 + k] = f2bf(v);
            }
        }
        __syncthreads();

        v16bf af = load_frag_f32(As, wm * 16, lane);
#pragma unroll
        for (int a = 0; a < ACC; ++a) {
            v16bf bf = load_frag(Bs, 32, (wn * ACC + a) * 16, 0, lane);
            acc[a] = __builtin_amdgcn_wmma_f32_16x16x32_bf16(
                false, af, false, bf, (short)0, acc[a], false, false);
        }
        __syncthreads();
    }

    // C layout: VGPR j, lanes 0-15 -> M=j ; lanes 16-31 -> M=j+8 ; N=lane&15
    const int ml = (lane < 16) ? 0 : 8;
    const int nl = lane & 15;
#pragma unroll
    for (int a = 0; a < ACC; ++a) {
#pragma unroll
        for (int j = 0; j < 8; ++j) {
            int r = m0 + wm * 16 + ml + j;
            int c = n0 + (wn * ACC + a) * 16 + nl;
            if (r < M && c < N) {
                float v = acc[a][j];
                if (bias) v += bias[c];
                if (act == 1)      v = v > 0.f ? v : 0.f;
                else if (act == 2) v = v > 0.f ? v : (expf(v) - 1.f);
                C[(long)r * ldc + c] = v;
            }
        }
    }
}

// ---------------------------------------------------------------------------
// Fused FF block: Y[M,64] = relu(X[M,64] @ W1[64,2048] + b1) @ W2[2048,64] + b2
// Never materializes the M x 2048 intermediate in HBM.
// ---------------------------------------------------------------------------
__global__ __launch_bounds__(256)
void ff_fused_k(const float* __restrict__ X, const float* __restrict__ W1p,
                const float* __restrict__ b1p, const float* __restrict__ W2p,
                const float* __restrict__ b2p, float* __restrict__ Y, int M) {
    __shared__ __attribute__((aligned(16))) unsigned short Xs[64 * 64];   // [row][k]
    __shared__ __attribute__((aligned(16))) unsigned short W1s[32 * 64];  // [ffcol][k]
    __shared__ __attribute__((aligned(16))) unsigned short Ts[64 * 32];   // [row][ffk]
    __shared__ __attribute__((aligned(16))) unsigned short W2s[64 * 32];  // [outcol][ffk]

    const int tid = threadIdx.x;
    const int lane = tid & 31;
    const int w = tid >> 5;
    const int wm = w & 3;      // row group (4 x 16 rows)
    const int wn = w >> 2;     // col group (2 x 32 out cols / 2 x 16 ff cols)
    const int r0 = blockIdx.x * 64;

    for (int i = tid; i < 64 * 64; i += 256) {
        int r = i >> 6, k = i & 63;
        int re = imin(r0 + r, M - 1);            // clamp: fringe rows unused
        Xs[i] = f2bf(X[(long)re * HID + k]);
    }

    v8f acc0 = {};
    v8f acc1 = {};
    const int ml = (lane < 16) ? 0 : 8;
    const int nl = lane & 15;

    for (int fc = 0; fc < FFDIM; fc += 32) {
        for (int i = tid; i < 32 * 64; i += 256) {         // W1[k][fc+n] -> [n][k]
            int n = i >> 6, k = i & 63;
            W1s[i] = f2bf(W1p[(long)k * FFDIM + fc + n]);
        }
        for (int i = tid; i < 64 * 32; i += 256) {         // W2[fc+k][n] -> [n][k]
            int n = i >> 5, k = i & 31;
            W2s[i] = f2bf(W2p[(long)(fc + k) * HID + n]);
        }
        __syncthreads();

        // stage 1: t(64x32) = relu(X @ W1chunk + b1)
        v8f t = {};
        {
            v16bf a0 = load_frag(Xs, 64, wm * 16, 0, lane);
            v16bf a1 = load_frag(Xs, 64, wm * 16, 32, lane);
            v16bf c0 = load_frag(W1s, 64, wn * 16, 0, lane);
            v16bf c1 = load_frag(W1s, 64, wn * 16, 32, lane);
            t = __builtin_amdgcn_wmma_f32_16x16x32_bf16(false, a0, false, c0, (short)0, t, false, false);
            t = __builtin_amdgcn_wmma_f32_16x16x32_bf16(false, a1, false, c1, (short)0, t, false, false);
        }
#pragma unroll
        for (int j = 0; j < 8; ++j) {
            int m = wm * 16 + ml + j;
            int n = wn * 16 + nl;
            float v = t[j] + b1p[fc + n];
            v = v > 0.f ? v : 0.f;
            Ts[m * 32 + n] = f2bf(v);
        }
        __syncthreads();

        // stage 2: acc(16x32 per wave) += Ts(16x32) @ W2chunk(32x32)
        {
            v16bf ta  = load_frag(Ts, 32, wm * 16, 0, lane);
            v16bf wb0 = load_frag(W2s, 32, wn * 32 + 0, 0, lane);
            v16bf wb1 = load_frag(W2s, 32, wn * 32 + 16, 0, lane);
            acc0 = __builtin_amdgcn_wmma_f32_16x16x32_bf16(false, ta, false, wb0, (short)0, acc0, false, false);
            acc1 = __builtin_amdgcn_wmma_f32_16x16x32_bf16(false, ta, false, wb1, (short)0, acc1, false, false);
        }
        __syncthreads();
    }

#pragma unroll
    for (int j = 0; j < 8; ++j) {
        int r = r0 + wm * 16 + ml + j;
        if (r < M) {
            int c0 = wn * 32 + nl;
            int c1 = wn * 32 + 16 + nl;
            Y[(long)r * HID + c0] = acc0[j] + b2p[c0];
            Y[(long)r * HID + c1] = acc1[j] + b2p[c1];
        }
    }
}

// ---------------------------------------------------------------------------
// GAT attention helpers
// ---------------------------------------------------------------------------
__global__ void fill_f32_k(float* p, float v, long n) {
    long i = (long)blockIdx.x * blockDim.x + threadIdx.x;
    if (i < n) p[i] = v;
}

__global__ void gat_logits_k(const float* __restrict__ h, const float* __restrict__ a_s,
                             const float* __restrict__ a_d, float* __restrict__ als,
                             float* __restrict__ ald, int Nn, int Hh, int D) {
    long i = (long)blockIdx.x * blockDim.x + threadIdx.x;
    if (i >= (long)Nn * Hh) return;
    int n = (int)(i / Hh), hd = (int)(i % Hh);
    const float* hp = h + ((long)n * Hh + hd) * D;
    float ss = 0.f, sd = 0.f;
    for (int d = 0; d < D; ++d) {
        float v = hp[d];
        ss += v * a_s[hd * D + d];
        sd += v * a_d[hd * D + d];
    }
    als[i] = ss;
    ald[i] = sd;
}

__global__ void edge_max_k(const float* __restrict__ als, const float* __restrict__ ald,
                           const int* __restrict__ esrc, const int* __restrict__ edst,
                           float* __restrict__ ebuf, float* __restrict__ mb,
                           int E_, int N_, int H_) {
    long i = (long)blockIdx.x * blockDim.x + threadIdx.x;
    long tot = (long)(E_ + N_) * H_;
    if (i >= tot) return;
    int eid = (int)(i / H_), h = (int)(i % H_);
    int s = eid < E_ ? esrc[eid] : (eid - E_);
    int d = eid < E_ ? edst[eid] : (eid - E_);
    float e = als[(long)s * H_ + h] + ald[(long)d * H_ + h];
    e = e > 0.f ? e : 0.2f * e;
    ebuf[i] = e;
    float* addr = mb + (long)d * H_ + h;
    if (e >= 0.f) atomicMax((int*)addr, __float_as_int(e));
    else          atomicMin((unsigned int*)addr, __float_as_uint(e));
}

__global__ void edge_sum_k(const int* __restrict__ esrc, const int* __restrict__ edst,
                           const float* __restrict__ mb, float* __restrict__ ebuf,
                           float* __restrict__ sb, int E_, int N_, int H_) {
    long i = (long)blockIdx.x * blockDim.x + threadIdx.x;
    long tot = (long)(E_ + N_) * H_;
    if (i >= tot) return;
    int eid = (int)(i / H_), h = (int)(i % H_);
    int d = eid < E_ ? edst[eid] : (eid - E_);
    float ex = expf(ebuf[i] - mb[(long)d * H_ + h]);
    ebuf[i] = ex;
    atomicAdd(&sb[(long)d * H_ + h], ex);
}

__global__ void edge_scatter_k(const float* __restrict__ hlin, const float* __restrict__ ebuf,
                               const float* __restrict__ sb, const int* __restrict__ esrc,
                               const int* __restrict__ edst, float* __restrict__ outp,
                               int E_, int N_, int H_, int D_) {
    long i = (long)blockIdx.x * blockDim.x + threadIdx.x;
    int HD = H_ * D_;
    long tot = (long)(E_ + N_) * HD;
    if (i >= tot) return;
    int eid = (int)(i / HD);
    int rem = (int)(i % HD);
    int h = rem / D_;
    int s = eid < E_ ? esrc[eid] : (eid - E_);
    int d = eid < E_ ? edst[eid] : (eid - E_);
    float alpha = ebuf[(long)eid * H_ + h] / (sb[(long)d * H_ + h] + 1e-16f);
    atomicAdd(&outp[(long)d * HD + rem], hlin[(long)s * HD + rem] * alpha);
}

__global__ void bias_act_k(float* __restrict__ p, const float* __restrict__ bias,
                           long n, int C, int act) {
    long i = (long)blockIdx.x * blockDim.x + threadIdx.x;
    if (i >= n) return;
    float v = p[i] + bias[(int)(i % C)];
    if (act == 2) v = v > 0.f ? v : (expf(v) - 1.f);
    p[i] = v;
}

__global__ void add_ln_k(const float* __restrict__ x, const float* __restrict__ r,
                         const float* __restrict__ g, const float* __restrict__ b,
                         float* __restrict__ y, int Nn) {
    int node = (int)(((long)blockIdx.x * blockDim.x + threadIdx.x) >> 5);
    int lane = threadIdx.x & 31;
    if (node >= Nn) return;
    long base = (long)node * HID;
    float v0 = x[base + lane] + r[base + lane];
    float v1 = x[base + 32 + lane] + r[base + 32 + lane];
    float s = v0 + v1;
    for (int o = 16; o > 0; o >>= 1) s += __shfl_xor(s, o, 32);
    float mu = s * (1.f / 64.f);
    float d0 = v0 - mu, d1 = v1 - mu;
    float q = d0 * d0 + d1 * d1;
    for (int o = 16; o > 0; o >>= 1) q += __shfl_xor(q, o, 32);
    float rs = rsqrtf(q * (1.f / 64.f) + 1e-5f);
    y[base + lane]      = d0 * rs * g[lane]      + b[lane];
    y[base + 32 + lane] = d1 * rs * g[lane + 32] + b[lane + 32];
}

// ---------------------------------------------------------------------------
// Orchestration
// ---------------------------------------------------------------------------
template <int BM, int BN>
static inline void launch_gemm(const float* A, const float* B, float* C, const float* bias,
                               int M, int N, int K, int lda, int ldb, int ldc, int act,
                               hipStream_t s) {
    dim3 grid(CDIV(N, BN), CDIV(M, BM));
    gemm_bf16_k<BM, BN><<<grid, 256, 0, s>>>(A, B, C, bias, M, N, K, lda, ldb, ldc, act);
}

static inline void launch_fill(float* p, float v, long n, hipStream_t s) {
    fill_f32_k<<<(int)CDIV(n, 256), 256, 0, s>>>(p, v, n);
}

static void run_gat_attn(const float* hlin, const float* a_s, const float* a_d,
                         const int* esrc, const int* edst,
                         float* als, float* ald, float* mb, float* sb, float* ebuf,
                         float* outp, int H, int D, hipStream_t s) {
    long nh = (long)NN * H;
    gat_logits_k<<<(int)CDIV(nh, 256), 256, 0, s>>>(hlin, a_s, a_d, als, ald, NN, H, D);
    launch_fill(mb, -INFINITY, nh, s);
    launch_fill(sb, 0.f, nh, s);
    long eh = (long)(EE + NN) * H;
    edge_max_k<<<(int)CDIV(eh, 256), 256, 0, s>>>(als, ald, esrc, edst, ebuf, mb, EE, NN, H);
    edge_sum_k<<<(int)CDIV(eh, 256), 256, 0, s>>>(esrc, edst, mb, ebuf, sb, EE, NN, H);
    long ehd = (long)(EE + NN) * H * D;
    edge_scatter_k<<<(int)CDIV(ehd, 256), 256, 0, s>>>(hlin, ebuf, sb, esrc, edst, outp, EE, NN, H, D);
}

extern "C" void kernel_launch(void* const* d_in, const int* in_sizes, int n_in,
                              void* d_out, int out_size, void* d_ws, size_t ws_size,
                              hipStream_t stream) {
    (void)in_sizes; (void)n_in; (void)out_size; (void)ws_size;

    const float* x      = (const float*)d_in[0];
    const int*   esrc   = (const int*)d_in[1];
    const int*   edst   = (const int*)d_in[2];
    const float* W1     = (const float*)d_in[3];
    const float* a_s1   = (const float*)d_in[4];
    const float* a_d1   = (const float*)d_in[5];
    const float* b1     = (const float*)d_in[6];
    const float* W2     = (const float*)d_in[7];
    const float* a_s2   = (const float*)d_in[8];
    const float* a_d2   = (const float*)d_in[9];
    const float* b2     = (const float*)d_in[10];
    // d_in[11] = t_Wq, d_in[12] = t_Wk : dead code (softmax over length-1 seq)
    const float* t_Wv   = (const float*)d_in[13];
    const float* t_bv   = (const float*)d_in[16];
    const float* t_Wo   = (const float*)d_in[17];
    const float* t_bo   = (const float*)d_in[18];
    const float* t_ln1g = (const float*)d_in[19];
    const float* t_ln1b = (const float*)d_in[20];
    const float* t_Wff1 = (const float*)d_in[21];
    const float* t_bff1 = (const float*)d_in[22];
    const float* t_Wff2 = (const float*)d_in[23];
    const float* t_bff2 = (const float*)d_in[24];
    const float* t_ln2g = (const float*)d_in[25];
    const float* t_ln2b = (const float*)d_in[26];
    const float* W3     = (const float*)d_in[27];
    const float* a_s3   = (const float*)d_in[28];
    const float* a_d3   = (const float*)d_in[29];
    const float* b3     = (const float*)d_in[30];

    // ---- workspace layout (f32) ----
    float* p = (float*)d_ws;
    float* h1lin = p;  p += (long)NN * 256;
    float* h1out = p;  p += (long)NN * 256;
    float* als   = p;  p += (long)NN * HEADS;
    float* ald   = p;  p += (long)NN * HEADS;
    float* mb    = p;  p += (long)NN * HEADS;
    float* sb    = p;  p += (long)NN * HEADS;
    float* ebuf  = p;  p += (long)(EE + NN) * HEADS;
    float* h2lin = p;  p += (long)NN * HID;
    float* rna   = p;  p += (long)NN * HID;
    float* t0    = p;  p += (long)NN * HID;
    float* t1    = p;  p += (long)NN * HID;
    float* t2    = p;  p += (long)NN * HID;
    float* t3    = p;  p += (long)NN * HID;
    float* xb1   = p;  p += (long)NN * HID;
    float* xb2   = p;  p += (long)NN * HID;
    float* h3lin = p;  p += (long)NN * OUT_DIM;

    // ================= GAT layer 1 (H=4, D=64) =================
    launch_gemm<32, 256>(x, W1, h1lin, nullptr, NN, HEADS * HID, IN_DIM,
                         IN_DIM, HEADS * HID, HEADS * HID, 0, stream);
    launch_fill(h1out, 0.f, (long)NN * 256, stream);
    run_gat_attn(h1lin, a_s1, a_d1, esrc, edst, als, ald, mb, sb, ebuf,
                 h1out, HEADS, HID, stream);
    bias_act_k<<<(int)CDIV((long)NN * 256, 256), 256, 0, stream>>>(
        h1out, b1, (long)NN * 256, 256, 2 /*elu*/);

    // ================= GAT layer 2 (H=1, D=64) =================
    launch_gemm<64, 64>(h1out, W2, h2lin, nullptr, NN, HID, HEADS * HID,
                        HEADS * HID, HID, HID, 0, stream);
    launch_fill(rna, 0.f, (long)NN * HID, stream);
    run_gat_attn(h2lin, a_s2, a_d2, esrc, edst, als, ald, mb, sb, ebuf,
                 rna, 1, HID, stream);
    bias_act_k<<<(int)CDIV((long)NN * HID, 256), 256, 0, stream>>>(
        rna, b2, (long)NN * HID, HID, 2 /*elu*/);

    // ================= Transformer fusion (seq_len=1) =================
    const float* xin = rna;
    float* xout[2] = {xb1, xb2};
    for (int l = 0; l < NLAYERS; ++l) {
        const long wOff  = (long)l * HID * HID;
        const long bOff  = (long)l * HID;
        // attn = (x @ Wv + bv) @ Wo + bo   (softmax over length-1 seq == 1)
        launch_gemm<64, 64>(xin, t_Wv + wOff, t0, t_bv + bOff, NN, HID, HID, HID, HID, HID, 0, stream);
        launch_gemm<64, 64>(t0, t_Wo + wOff, t1, t_bo + bOff, NN, HID, HID, HID, HID, HID, 0, stream);
        add_ln_k<<<CDIV(NN * 32, 256), 256, 0, stream>>>(
            xin, t1, t_ln1g + bOff, t_ln1b + bOff, t2, NN);
        ff_fused_k<<<CDIV(NN, 64), 256, 0, stream>>>(
            t2, t_Wff1 + (long)l * HID * FFDIM, t_bff1 + (long)l * FFDIM,
            t_Wff2 + (long)l * FFDIM * HID, t_bff2 + bOff, t3, NN);
        add_ln_k<<<CDIV(NN * 32, 256), 256, 0, stream>>>(
            t2, t3, t_ln2g + bOff, t_ln2b + bOff, xout[l], NN);
        xin = xout[l];
    }

    // ================= GAT head (H=1, D=25) =================
    float* adt = (float*)d_out;                       // [N, 25]
    launch_gemm<64, 32>(xin, W3, h3lin, nullptr, NN, OUT_DIM, HID,
                        HID, OUT_DIM, OUT_DIM, 0, stream);
    launch_fill(adt, 0.f, (long)NN * OUT_DIM, stream);
    run_gat_attn(h3lin, a_s3, a_d3, esrc, edst, als, ald, mb, sb, ebuf,
                 adt, 1, OUT_DIM, stream);
    bias_act_k<<<(int)CDIV((long)NN * OUT_DIM, 256), 256, 0, stream>>>(
        adt, b3, (long)NN * OUT_DIM, OUT_DIM, 0 /*none*/);

    // fused output
    hipMemcpyAsync((float*)d_out + (long)NN * OUT_DIM, xin,
                   (size_t)NN * HID * sizeof(float),
                   hipMemcpyDeviceToDevice, stream);
}
